// DecoderLayer_16458314678686
// MI455X (gfx1250) — matrix operand order repaired and checked by reading, not compiled, not run on hardware
//
#include <hip/hip_runtime.h>
#include <hip/hip_bf16.h>

// ---------------------------------------------------------------------------
// Types for CDNA5 WMMA (gfx1250, wave32): D(f32 16x16) = A(bf16 16x32) x B(bf16 32x16) + C
// ---------------------------------------------------------------------------
typedef __attribute__((ext_vector_type(16))) __bf16 v16bf;
typedef __attribute__((ext_vector_type(8)))  float  v8f;

union BF16Frag {
    v16bf v;
    uint4 q[2];
};

__device__ __forceinline__ v8f wmma_bf16(v16bf a, v16bf b, v8f c) {
    // 8 args: (neg_a, A, neg_b, B, c_mod, C, reuse_a, reuse_b)
    return __builtin_amdgcn_wmma_f32_16x16x32_bf16(false, a, false, b, (short)0, c, false, false);
}

// ---------------------------------------------------------------------------
// CDNA5 async Global->LDS copy via inline asm (GLOBAL_LOAD_ASYNC_TO_LDS_B128,
// tracked by ASYNCcnt; ISA §15.18 / §10.7). Operands: VDST = LDS address VGPR,
// VADDR = 64-bit global address, "off" = no SADDR. INST_OFFSET is added to
// BOTH the LDS and global addresses (ISA §10.7.4 pseudocode).
// ---------------------------------------------------------------------------
typedef __attribute__((address_space(3))) __bf16 as3_bf16;

__device__ __forceinline__ void async_copy_b128(const __bf16* g, __bf16* l) {
    as3_bf16* l3 = (as3_bf16*)l;
    asm volatile("global_load_async_to_lds_b128 %0, %1, off"
                 :: "v"(l3), "v"(g) : "memory");
}

// Copy 32 bytes: two b128 async ops sharing one address, second at offset 16
// (offset applies to both LDS and global sides).
__device__ __forceinline__ void async_copy_b256(const __bf16* g, __bf16* l) {
    as3_bf16* l3 = (as3_bf16*)l;
    asm volatile("global_load_async_to_lds_b128 %0, %1, off"
                 :: "v"(l3), "v"(g) : "memory");
    asm volatile("global_load_async_to_lds_b128 %0, %1, off offset:16"
                 :: "v"(l3), "v"(g) : "memory");
}

__device__ __forceinline__ void wait_async_copies() {
#if __has_builtin(__builtin_amdgcn_s_wait_asynccnt)
    __builtin_amdgcn_s_wait_asynccnt(0);
#else
    asm volatile("s_wait_asynccnt 0x0" ::: "memory");
#endif
}

// ---------------------------------------------------------------------------
// fp32 -> bf16 conversion (grid-stride)
// ---------------------------------------------------------------------------
__global__ void f32_to_bf16_kernel(const float* __restrict__ in,
                                   __bf16* __restrict__ out, size_t n) {
    size_t i = (size_t)blockIdx.x * blockDim.x + threadIdx.x;
    const size_t stride = (size_t)gridDim.x * blockDim.x;
    for (; i < n; i += stride) out[i] = (__bf16)in[i];
}

// ---------------------------------------------------------------------------
// Tiled bf16 GEMM: C[M,N] = A[M,K] (row-major bf16) x B[K,N] (row-major bf16)
// Block tile 128x128, K-step 32, double-buffered LDS (1 barrier/iter).
// A tile: async Global->LDS DMA issued for tile t+1 while tile t computes.
// B tile: register-staged + transposed into LDS [n][k] after the WMMA block.
// 8 waves, each computes 32x64 (2x4 WMMA tiles). M,N,K multiples of 128.
// ---------------------------------------------------------------------------
#define LDSW 40  // padded LDS row stride in halfwords (80 B, 16B aligned)

__global__ __launch_bounds__(256)
void gemm_bf16_kernel(const __bf16* __restrict__ A, const __bf16* __restrict__ B,
                      const float* __restrict__ bias,
                      float* __restrict__ outF, __bf16* __restrict__ outH,
                      int M, int N, int K, int relu) {
    __shared__ __align__(16) __bf16 Alds[2][128 * LDSW];   // [m][k] row-major
    __shared__ __align__(16) __bf16 Blds[2][128 * LDSW];   // transposed: [n][k]

    const int tid  = threadIdx.x;
    const int lane = tid & 31;
    const int wid  = tid >> 5;
    const int l16  = lane & 15;
    const int hi8  = (lane >> 4) << 3;   // 0 or 8
    const int hi16 = (lane >> 4) << 4;   // 0 or 16

    const int bm = blockIdx.y * 128;
    const int bn = blockIdx.x * 128;
    const int wm = (wid >> 1) * 32;      // wave row offset within block
    const int wn = (wid & 1) * 64;       // wave col offset within block

    const int ar  = tid >> 2;            // A stage row (0..63), +64 second pass
    const int aq  = tid & 3;             // A stage 8-elem quad
    const int bk  = tid >> 3;            // B stage k row (0..31)
    const int bnb = (tid & 7) * 16;      // B stage n base

    auto stageA = [&](int ks, int buf) {
        async_copy_b128(A + (size_t)(bm + ar) * K + ks + aq * 8,
                        &Alds[buf][ar * LDSW + aq * 8]);
        async_copy_b128(A + (size_t)(bm + ar + 64) * K + ks + aq * 8,
                        &Alds[buf][(ar + 64) * LDSW + aq * 8]);
    };
    auto loadB = [&](int ks, uint4& p0, uint4& p1) {
        const uint4* s = (const uint4*)(B + (size_t)(ks + bk) * N + bn + bnb);
        p0 = s[0];
        p1 = s[1];
    };
    auto storeBT = [&](int buf, uint4 p0, uint4 p1) {
        const __bf16* e0 = (const __bf16*)&p0;
        const __bf16* e1 = (const __bf16*)&p1;
        #pragma unroll
        for (int j = 0; j < 8; ++j) Blds[buf][(bnb + j) * LDSW + bk] = e0[j];
        #pragma unroll
        for (int j = 0; j < 8; ++j) Blds[buf][(bnb + 8 + j) * LDSW + bk] = e1[j];
    };

    v8f acc[2][4] = {};
    uint4 bs0, bs1;

    // Prologue: stage tile 0 into buffer 0
    stageA(0, 0);
    loadB(0, bs0, bs1);
    storeBT(0, bs0, bs1);
    wait_async_copies();
    __syncthreads();

    const int nt = K / 32;
    for (int i = 0; i < nt; ++i) {
        const int cur = i & 1, nxt = cur ^ 1;
        const int ks  = (i + 1 < nt ? i + 1 : i) * 32;  // clamp; redundant last reload

        // Issue next tile's copies early (hidden behind the WMMA block)
        stageA(ks, nxt);
        loadB(ks, bs0, bs1);
        __builtin_prefetch(B + (size_t)(ks + bk) * N + bn + bnb, 0, 3);

        // Fragments per ISA layouts:
        //  A elem i: m=lane&15, k=(i/8)*16 + hi8 + (i%8)  -> 2x b128 from row-major LDS
        //  B elem i: n=lane&15, k=(i/8)*16 + hi8 + (i%8)  -> 2x b128 from transposed LDS
        BF16Frag af[2], bfr[4];
        #pragma unroll
        for (int t = 0; t < 2; ++t) {
            const int r = wm + t * 16 + l16;
            af[t].q[0] = *(const uint4*)&Alds[cur][r * LDSW + hi8];
            af[t].q[1] = *(const uint4*)&Alds[cur][r * LDSW + 16 + hi8];
        }
        #pragma unroll
        for (int j = 0; j < 4; ++j) {
            const int n = wn + j * 16 + l16;
            bfr[j].q[0] = *(const uint4*)&Blds[cur][n * LDSW + hi8];
            bfr[j].q[1] = *(const uint4*)&Blds[cur][n * LDSW + 16 + hi8];
        }
        #pragma unroll
        for (int t = 0; t < 2; ++t)
            #pragma unroll
            for (int j = 0; j < 4; ++j)
                acc[t][j] = wmma_bf16(af[t].v, bfr[j].v, acc[t][j]);

        // Finish staging tile t+1, publish, single barrier
        storeBT(nxt, bs0, bs1);
        wait_async_copies();
        __syncthreads();
    }

    // Epilogue: C elem r -> m = r + hi8, n = lane&15
    #pragma unroll
    for (int t = 0; t < 2; ++t) {
        #pragma unroll
        for (int j = 0; j < 4; ++j) {
            const int col = bn + wn + j * 16 + l16;
            const float bv = bias ? bias[col] : 0.0f;
            #pragma unroll
            for (int r = 0; r < 8; ++r) {
                const int row = bm + wm + t * 16 + r + hi8;
                float v = acc[t][j][r] + bv;
                if (relu) v = fmaxf(v, 0.0f);
                const size_t idx = (size_t)row * N + col;
                if (outF) outF[idx] = v;
                if (outH) outH[idx] = (__bf16)v;
            }
        }
    }
}

// ---------------------------------------------------------------------------
// Flash attention: one block = 128 query rows of one (batch, head); 8 waves,
// each wave owns 16 query rows. Double-buffered 64-key tiles: K via async
// Global->LDS DMA ([key][dh]), V register-staged + transposed ([dh][key]).
// S = Q K^T via WMMA, online softmax in fp32 (shfl_xor width-16 reductions),
// P restaged per-wave in LDS in A-frag layout, O += P V via WMMA.
// ---------------------------------------------------------------------------
__global__ __launch_bounds__(256)
void flash_attn_kernel(const __bf16* __restrict__ Q, const __bf16* __restrict__ Kb,
                       const __bf16* __restrict__ Vb, const float* __restrict__ mask,
                       __bf16* __restrict__ ctx, int LK, int TGTc) {
    __shared__ __align__(16) __bf16 Klds[2][64 * 72];    // [key][dh]
    __shared__ __align__(16) __bf16 VT[2][64 * 72];      // [dh][key]
    __shared__ __align__(16) __bf16 Plds[8][16 * 72];    // per-wave P staging [m][key]

    const int tid  = threadIdx.x;
    const int lane = tid & 31;
    const int wid  = tid >> 5;
    const int l16  = lane & 15;
    const int hi8  = (lane >> 4) << 3;
    const int hi16 = (lane >> 4) << 4;

    const int bh = blockIdx.y;
    const int b  = bh >> 4;              // N_HEADS = 16
    const int h  = bh & 15;
    const int q0 = blockIdx.x * 128 + wid * 16;

    // Q fragments for this wave's 16 rows: 2 chunks of dh (0..31, 32..63)
    BF16Frag qa[2];
    {
        const __bf16* qp = Q + ((size_t)(b * TGTc + q0 + l16)) * 1024 + h * 64;
        #pragma unroll
        for (int c = 0; c < 2; ++c) {
            qa[c].q[0] = *(const uint4*)(qp + c * 32 + hi8);
            qa[c].q[1] = *(const uint4*)(qp + c * 32 + 16 + hi8);
        }
    }

    float mrow[8], lrow[8];
    #pragma unroll
    for (int r = 0; r < 8; ++r) { mrow[r] = -1e30f; lrow[r] = 0.0f; }
    v8f o[4] = {};

    const __bf16* kbase = Kb + (size_t)b * LK * 1024 + h * 64;
    const __bf16* vbase = Vb + (size_t)b * LK * 1024 + h * 64;
    const float scale = 0.125f;  // 1/sqrt(64)

    const int key = tid >> 2;            // 0..63
    const int d0  = (tid & 3) * 16;      // dh base for staging

    auto stageK = [&](int kt, int buf) {
        async_copy_b256(kbase + (size_t)(kt + key) * 1024 + d0,
                        &Klds[buf][key * 72 + d0]);
    };
    auto loadV = [&](int kt, uint4& p0, uint4& p1) {
        const uint4* vs = (const uint4*)(vbase + (size_t)(kt + key) * 1024 + d0);
        p0 = vs[0];
        p1 = vs[1];
    };
    auto storeVT = [&](int buf, uint4 p0, uint4 p1) {
        const __bf16* e0 = (const __bf16*)&p0;
        const __bf16* e1 = (const __bf16*)&p1;
        #pragma unroll
        for (int j = 0; j < 8; ++j) VT[buf][(d0 + j) * 72 + key]     = e0[j];
        #pragma unroll
        for (int j = 0; j < 8; ++j) VT[buf][(d0 + 8 + j) * 72 + key] = e1[j];
    };

    uint4 vs0, vs1;
    stageK(0, 0);
    loadV(0, vs0, vs1);
    storeVT(0, vs0, vs1);
    wait_async_copies();
    __syncthreads();

    const int ntk = LK / 64;
    for (int i = 0; i < ntk; ++i) {
        const int cur = i & 1, nxt = cur ^ 1;
        const int kt  = i * 64;
        const int ktn = (i + 1 < ntk ? i + 1 : i) * 64;  // clamp; redundant last reload

        // Issue next tile's copies early
        stageK(ktn, nxt);
        loadV(ktn, vs0, vs1);

        // S tiles (16 q-rows x 64 keys = 4 tiles), 2 dh-chunks each
        v8f s[4] = {};
        #pragma unroll
        for (int j = 0; j < 4; ++j) {
            #pragma unroll
            for (int c = 0; c < 2; ++c) {
                BF16Frag kf;
                const int n = j * 16 + l16;  // key within tile
                kf.q[0] = *(const uint4*)&Klds[cur][n * 72 + c * 32 + hi8];
                kf.q[1] = *(const uint4*)&Klds[cur][n * 72 + c * 32 + 16 + hi8];
                s[j] = wmma_bf16(qa[c].v, kf.v, s[j]);
            }
        }
        // scale + additive mask
        #pragma unroll
        for (int j = 0; j < 4; ++j) {
            const int kc = kt + j * 16 + l16;
            #pragma unroll
            for (int r = 0; r < 8; ++r) {
                const int qr = q0 + r + hi8;
                s[j][r] = s[j][r] * scale + mask[((size_t)(b * TGTc + qr)) * LK + kc];
            }
        }
        // online softmax per query row (rows split across lane halves by C layout)
        #pragma unroll
        for (int r = 0; r < 8; ++r) {
            float mx = fmaxf(fmaxf(s[0][r], s[1][r]), fmaxf(s[2][r], s[3][r]));
            #pragma unroll
            for (int m = 1; m < 16; m <<= 1) mx = fmaxf(mx, __shfl_xor(mx, m, 16));
            const float mnew = fmaxf(mrow[r], mx);
            const float corr = __expf(mrow[r] - mnew);
            float ps = 0.0f;
            #pragma unroll
            for (int j = 0; j < 4; ++j) {
                const float pv = __expf(s[j][r] - mnew);
                s[j][r] = pv;
                ps += pv;
            }
            #pragma unroll
            for (int m = 1; m < 16; m <<= 1) ps += __shfl_xor(ps, m, 16);
            lrow[r] = lrow[r] * corr + ps;
            mrow[r] = mnew;
            #pragma unroll
            for (int d = 0; d < 4; ++d) o[d][r] *= corr;
        }
        // stash P (bf16) into per-wave LDS (intra-wave: DS ops stay in order)
        #pragma unroll
        for (int j = 0; j < 4; ++j)
            #pragma unroll
            for (int r = 0; r < 8; ++r)
                Plds[wid][(r + hi8) * 72 + j * 16 + l16] = (__bf16)s[j][r];
        // O += P x V   (A = P from LDS in A-frag layout, B = V^T fragments)
        #pragma unroll
        for (int c = 0; c < 2; ++c) {
            BF16Frag pf;
            pf.q[0] = *(const uint4*)&Plds[wid][l16 * 72 + c * 32 + hi8];
            pf.q[1] = *(const uint4*)&Plds[wid][l16 * 72 + c * 32 + 16 + hi8];
            #pragma unroll
            for (int d = 0; d < 4; ++d) {
                BF16Frag vf;
                const int n = d * 16 + l16;  // dh column
                vf.q[0] = *(const uint4*)&VT[cur][n * 72 + c * 32 + hi8];
                vf.q[1] = *(const uint4*)&VT[cur][n * 72 + c * 32 + 16 + hi8];
                o[d] = wmma_bf16(pf.v, vf.v, o[d]);
            }
        }

        // Finish staging tile t+1, publish, single barrier
        storeVT(nxt, vs0, vs1);
        wait_async_copies();
        __syncthreads();
    }

    // normalize + store context (bf16, [B*TGT, 1024] with head at h*64)
    #pragma unroll
    for (int d = 0; d < 4; ++d) {
        #pragma unroll
        for (int r = 0; r < 8; ++r) {
            const float val = o[d][r] / lrow[r];
            const int qr  = q0 + r + hi8;
            const int col = h * 64 + d * 16 + l16;
            ctx[((size_t)(b * TGTc + qr)) * 1024 + col] = (__bf16)val;
        }
    }
}

// ---------------------------------------------------------------------------
// Fused residual add + LayerNorm over 1024 columns. One 256-thread block per row.
// Writes fp32 (residual chain / final output) and optional bf16 (next GEMM input).
// ---------------------------------------------------------------------------
__global__ __launch_bounds__(256)
void add_layernorm_kernel(const float* __restrict__ y, const float* __restrict__ res,
                          const float* __restrict__ g, const float* __restrict__ beta,
                          float* __restrict__ outF, __bf16* __restrict__ outH) {
    constexpr int N = 1024;
    const int row = blockIdx.x, tid = threadIdx.x;
    const int lane = tid & 31, wid = tid >> 5;
    __shared__ float red[16];

    const float4 yv = ((const float4*)(y + (size_t)row * N))[tid];
    const float4 rv = ((const float4*)(res + (size_t)row * N))[tid];
    float x[4] = { yv.x + rv.x, yv.y + rv.y, yv.z + rv.z, yv.w + rv.w };

    float s = x[0] + x[1] + x[2] + x[3];
    #pragma unroll
    for (int m = 1; m < 32; m <<= 1) s += __shfl_xor(s, m, 32);
    if (lane == 0) red[wid] = s;
    __syncthreads();
    float tot = 0.0f;
    #pragma unroll
    for (int i = 0; i < 8; ++i) tot += red[i];
    const float mean = tot * (1.0f / N);

    float vs = 0.0f;
    #pragma unroll
    for (int i = 0; i < 4; ++i) { const float d = x[i] - mean; vs += d * d; }
    #pragma unroll
    for (int m = 1; m < 32; m <<= 1) vs += __shfl_xor(vs, m, 32);
    if (lane == 0) red[8 + wid] = vs;
    __syncthreads();
    float vtot = 0.0f;
    #pragma unroll
    for (int i = 0; i < 8; ++i) vtot += red[8 + i];
    const float rstd = rsqrtf(vtot * (1.0f / N) + 1e-5f);

    const float4 gv = ((const float4*)g)[tid];
    const float4 bv = ((const float4*)beta)[tid];
    const float gg[4] = { gv.x, gv.y, gv.z, gv.w };
    const float bb[4] = { bv.x, bv.y, bv.z, bv.w };
    #pragma unroll
    for (int i = 0; i < 4; ++i) {
        const float ov = (x[i] - mean) * rstd * gg[i] + bb[i];
        const size_t idx = (size_t)row * N + tid * 4 + i;
        if (outF) outF[idx] = ov;
        if (outH) outH[idx] = (__bf16)ov;
    }
}

// ---------------------------------------------------------------------------
// Host orchestration
// ---------------------------------------------------------------------------
extern "C" void kernel_launch(void* const* d_in, const int* in_sizes, int n_in,
                              void* d_out, int out_size, void* d_ws, size_t ws_size,
                              hipStream_t stream) {
    (void)in_sizes; (void)n_in; (void)out_size; (void)ws_size;

    constexpr int DM = 1024, DFF = 4096, NB = 4, TGT = 1024, SRC = 2048, NH = 16;
    constexpr int M1 = NB * TGT;   // 4096 decoder tokens
    constexpr int M2 = NB * SRC;   // 8192 encoder tokens

    const float* xf     = (const float*)d_in[0];
    const float* encf   = (const float*)d_in[1];
    const float* mself  = (const float*)d_in[2];
    const float* mcross = (const float*)d_in[3];
    const float* w_sa[4] = { (const float*)d_in[4], (const float*)d_in[5],
                             (const float*)d_in[6], (const float*)d_in[7] };
    const float* w_ca[4] = { (const float*)d_in[8], (const float*)d_in[9],
                             (const float*)d_in[10], (const float*)d_in[11] };
    const float* w1 = (const float*)d_in[12];
    const float* b1 = (const float*)d_in[13];
    const float* w2 = (const float*)d_in[14];
    const float* b2 = (const float*)d_in[15];
    const float* ln1g = (const float*)d_in[16]; const float* ln1b = (const float*)d_in[17];
    const float* ln2g = (const float*)d_in[18]; const float* ln2b = (const float*)d_in[19];
    const float* ln3g = (const float*)d_in[20]; const float* ln3b = (const float*)d_in[21];

    // Deterministic workspace carve-out (same layout every call)
    char* wp = (char*)d_ws;
    auto carve = [&](size_t elems, size_t esz) -> void* {
        void* r = (void*)wp;
        wp += ((elems * esz) + 255) & ~(size_t)255;
        return r;
    };
    __bf16* xb    = (__bf16*)carve((size_t)M1 * DM, 2);
    __bf16* encb  = (__bf16*)carve((size_t)M2 * DM, 2);
    __bf16* wsab[4]; for (int i = 0; i < 4; ++i) wsab[i] = (__bf16*)carve((size_t)DM * DM, 2);
    __bf16* wcab[4]; for (int i = 0; i < 4; ++i) wcab[i] = (__bf16*)carve((size_t)DM * DM, 2);
    __bf16* w1b   = (__bf16*)carve((size_t)DM * DFF, 2);
    __bf16* w2b   = (__bf16*)carve((size_t)DFF * DM, 2);
    __bf16* qb    = (__bf16*)carve((size_t)M1 * DM, 2);
    __bf16* kb    = (__bf16*)carve((size_t)M1 * DM, 2);
    __bf16* vb    = (__bf16*)carve((size_t)M1 * DM, 2);
    __bf16* ctxb  = (__bf16*)carve((size_t)M1 * DM, 2);
    float*  saF   = (float*)carve((size_t)M1 * DM, 4);
    float*  out1F = (float*)carve((size_t)M1 * DM, 4);
    __bf16* out1b = (__bf16*)carve((size_t)M1 * DM, 2);
    __bf16* caqb  = (__bf16*)carve((size_t)M1 * DM, 2);
    __bf16* cakb  = (__bf16*)carve((size_t)M2 * DM, 2);
    __bf16* cavb  = (__bf16*)carve((size_t)M2 * DM, 2);
    __bf16* ctx2b = (__bf16*)carve((size_t)M1 * DM, 2);
    float*  caF   = (float*)carve((size_t)M1 * DM, 4);
    float*  out2F = (float*)carve((size_t)M1 * DM, 4);
    __bf16* out2b = (__bf16*)carve((size_t)M1 * DM, 2);
    __bf16* hidb  = (__bf16*)carve((size_t)M1 * DFF, 2);
    float*  ffnF  = (float*)carve((size_t)M1 * DM, 4);

    auto conv = [&](const float* src, __bf16* dst, size_t n) {
        f32_to_bf16_kernel<<<1024, 256, 0, stream>>>(src, dst, n);
    };
    auto gemm = [&](const __bf16* A, const __bf16* B, const float* bias,
                    float* oF, __bf16* oH, int M, int N, int K, int relu) {
        dim3 grid(N / 128, M / 128);
        gemm_bf16_kernel<<<grid, 256, 0, stream>>>(A, B, bias, oF, oH, M, N, K, relu);
    };

    // 0) downcast activations + weights
    conv(xf, xb, (size_t)M1 * DM);
    conv(encf, encb, (size_t)M2 * DM);
    for (int i = 0; i < 4; ++i) conv(w_sa[i], wsab[i], (size_t)DM * DM);
    for (int i = 0; i < 4; ++i) conv(w_ca[i], wcab[i], (size_t)DM * DM);
    conv(w1, w1b, (size_t)DM * DFF);
    conv(w2, w2b, (size_t)DFF * DM);

    // 1) self-attention
    gemm(xb, wsab[0], nullptr, nullptr, qb, M1, DM, DM, 0);
    gemm(xb, wsab[1], nullptr, nullptr, kb, M1, DM, DM, 0);
    gemm(xb, wsab[2], nullptr, nullptr, vb, M1, DM, DM, 0);
    flash_attn_kernel<<<dim3(TGT / 128, NB * NH), 256, 0, stream>>>(
        qb, kb, vb, mself, ctxb, TGT, TGT);
    gemm(ctxb, wsab[3], nullptr, saF, nullptr, M1, DM, DM, 0);
    add_layernorm_kernel<<<M1, 256, 0, stream>>>(saF, xf, ln1g, ln1b, out1F, out1b);

    // 2) cross-attention
    gemm(out1b, wcab[0], nullptr, nullptr, caqb, M1, DM, DM, 0);
    gemm(encb, wcab[1], nullptr, nullptr, cakb, M2, DM, DM, 0);
    gemm(encb, wcab[2], nullptr, nullptr, cavb, M2, DM, DM, 0);
    flash_attn_kernel<<<dim3(TGT / 128, NB * NH), 256, 0, stream>>>(
        caqb, cakb, cavb, mcross, ctx2b, SRC, TGT);
    gemm(ctx2b, wcab[3], nullptr, caF, nullptr, M1, DM, DM, 0);
    add_layernorm_kernel<<<M1, 256, 0, stream>>>(caF, out1F, ln2g, ln2b, out2F, out2b);

    // 3) FFN
    gemm(out2b, w1b, b1, nullptr, hidb, M1, DFF, DM, 1);   // + bias, ReLU
    gemm(hidb, w2b, b2, ffnF, nullptr, M1, DM, DFF, 0);    // + bias
    add_layernorm_kernel<<<M1, 256, 0, stream>>>(ffnF, out2F, ln3g, ln3b,
                                                 (float*)d_out, nullptr);
}
